// SimpleNSMModel_1580547975092
// MI455X (gfx1250) — matrix operand order repaired and checked by reading, not compiled, not run on hardware
//
#include <hip/hip_runtime.h>
#include <hip/hip_bf16.h>
#include <math.h>

// Sizes from the reference
#define NB   8
#define NT   1024
#define ND   128
#define NS   8
#define NOUT 32000

typedef __attribute__((ext_vector_type(16))) __bf16 v16bf;
typedef __attribute__((ext_vector_type(8)))  float  v8f;

union FragU { v16bf v; uint4 q[2]; unsigned short h[16]; };
union AccU  { v8f   v; float f[8]; };

__device__ __forceinline__ unsigned short f2bf(float x) {
  unsigned u = __builtin_bit_cast(unsigned, x);
  u = (u + 0x7FFFu + ((u >> 16) & 1u)) >> 16;   // RNE
  return (unsigned short)u;
}
__device__ __forceinline__ float sigmoidf_(float x) { return 1.0f / (1.0f + expf(-x)); }

// A-frag: 16x32 bf16 tile from row-major LDS (stride in halves, 8-half aligned).
// lanes 0-15: row=lane, K = kb+{0..15}; lanes 16-31: K = kb+16+{0..15}
__device__ __forceinline__ v16bf load_afrag(const unsigned short* base, int strideH, int kb, int lane) {
  int row = lane & 15, sel = lane >> 4;
  const unsigned short* p = base + row * strideH + kb + sel * 16;
  FragU f;
  f.q[0] = *(const uint4*)(p);
  f.q[1] = *(const uint4*)(p + 8);
  return f.v;
}
// B-frag: pre-swizzled lane-major frags, 512 halves (1KB) per (ntile,kstep)
__device__ __forceinline__ v16bf load_bfrag(const unsigned short* frags, int nt, int ks, int ksteps, int lane) {
  const unsigned short* p = frags + (((nt * ksteps + ks) << 9) | (lane << 4));
  FragU f;
  f.q[0] = *(const uint4*)(p);
  f.q[1] = *(const uint4*)(p + 8);
  return f.v;
}
__device__ __forceinline__ v8f wmma_bf16(v16bf a, v16bf b, v8f c) {
  return __builtin_amdgcn_wmma_f32_16x16x32_bf16(false, a, false, b, (short)0, c, false, false);
}

// ---------------- weight pack: fp32 [N,K] row-major -> bf16 WMMA B-frag order ----------------
// frag element i: nt=i/(ksteps*512); ks; lane=li/16; j=li%16
// n = nt*16 + (lane&15); k = ks*32 + (lane>>4)*16 + j; value = W[n,k]
__global__ void pack_frags(const float* __restrict__ s0, const float* __restrict__ s1,
                           const float* __restrict__ s2, int rows_per, int K, int total,
                           unsigned short* __restrict__ dst) {
  int ksteps = K >> 5;
  for (int i = blockIdx.x * blockDim.x + threadIdx.x; i < total; i += gridDim.x * blockDim.x) {
    int frag = i >> 9;
    int li   = i & 511;
    int nt = frag / ksteps, ks = frag % ksteps;
    int lane = li >> 4, j = li & 15;
    int n = nt * 16 + (lane & 15);
    int k = ks * 32 + ((lane >> 4) << 4) + j;
    int which = n / rows_per;
    const float* src = (which == 0) ? s0 : ((which == 1) ? s1 : s2);
    dst[i] = f2bf(src[(size_t)(n % rows_per) * K + k]);
  }
}

// ---------------- LDS layout (bytes) for the persistent scan kernel ----------------
enum : unsigned {
  OFF_WZR  = 0,        // 16nt x 8ks x 1KB = 131072
  OFF_WH   = 131072,   // 8 x 8 x 1KB     = 65536
  OFF_WC   = 196608,   // 8 x 8 x 1KB     = 65536
  OFF_COMB = 262144,   // 16 x 264 halves = 8448  (cols 0-127 states/r*s, 128-255 x_t/attn)
  OFF_ACT2 = 270592,   // 16 x 136 halves = 4352  (attention head-mix, input to Wo)
  OFF_ST   = 274944,   // 8x128 f32 = 4096
  OFF_Z    = 279040,
  OFF_Q    = 283136,
  OFF_K    = 287232,
  OFF_V    = 291328,
  OFF_AT   = 295424,   // attn = o@Wo^T+bo (f32)
  OFF_SC   = 299520,   // 4x8x8 f32 scores = 1024
  OFF_BIAS = 300544,   // 1024 f32 = 4096
  LDS_TOTAL= 304640
};
#define COMB_STR 264
#define ACT_STR  136

__global__ __launch_bounds__(256, 1) void nsm_scan(
    const int* __restrict__ xin, const float* __restrict__ emb,
    const float* __restrict__ init_states,
    const uint4* __restrict__ wpack,            // wzr|wh|wc frags, 262144 B contiguous
    const unsigned short* __restrict__ wqkvf,   // 24nt x 4ks frags (L2-streamed)
    const unsigned short* __restrict__ wof,     //  8nt x 4ks frags (L2-streamed)
    const float* __restrict__ pbz, const float* __restrict__ pbr,
    const float* __restrict__ pbh, const float* __restrict__ pbq,
    const float* __restrict__ pbk, const float* __restrict__ pbv,
    const float* __restrict__ pbo, const float* __restrict__ pbc,
    float* __restrict__ st_out)
{
  extern __shared__ char smem[];
  unsigned short* wzr  = (unsigned short*)(smem + OFF_WZR);
  unsigned short* whf  = (unsigned short*)(smem + OFF_WH);
  unsigned short* wcf  = (unsigned short*)(smem + OFF_WC);
  unsigned short* comb = (unsigned short*)(smem + OFF_COMB);
  unsigned short* act2 = (unsigned short*)(smem + OFF_ACT2);
  float* st   = (float*)(smem + OFF_ST);
  float* zb   = (float*)(smem + OFF_Z);
  float* qb   = (float*)(smem + OFF_Q);
  float* kb   = (float*)(smem + OFF_K);
  float* vb   = (float*)(smem + OFF_V);
  float* at   = (float*)(smem + OFF_AT);
  float* sc   = (float*)(smem + OFF_SC);
  float* bias = (float*)(smem + OFF_BIAS);

  const int tid  = threadIdx.x;
  const int lane = tid & 31;
  const int wave = tid >> 5;
  const int b    = blockIdx.x;

  // stage hot weights to LDS: 256 KB as 16384 x uint4
  for (int i = tid; i < 16384; i += 256) ((uint4*)smem)[i] = wpack[i];
  if (tid < 128) {
    bias[      tid] = pbz[tid];  bias[128 + tid] = pbr[tid];
    bias[256 + tid] = pbh[tid];
    bias[384 + tid] = pbq[tid];  bias[512 + tid] = pbk[tid];  bias[640 + tid] = pbv[tid];
    bias[768 + tid] = pbo[tid];  bias[896 + tid] = pbc[tid];
  }
  for (int i = tid; i < 1024; i += 256) {   // states0 (broadcast of init_states)
    float v = init_states[i];
    int r = i >> 7, c = i & 127;
    st[i] = v;
    comb[r * COMB_STR + c] = f2bf(v);
  }
  for (int i = tid; i < 2048; i += 256) {   // zero comb rows 8..15 (M-padding)
    int r = 8 + (i >> 8), c = i & 255;
    comb[r * COMB_STR + c] = 0;
  }
  for (int i = tid; i < 1024; i += 256) {   // zero act2 rows 8..15
    int r = 8 + (i >> 7), c = i & 127;
    act2[r * ACT_STR + c] = 0;
  }
  __syncthreads();

  for (int t = 0; t < NT; ++t) {
    // ---- A: token embedding row -> comb[:,128:256]
    int tok = xin[b * NT + t];
    if (tid < 128) {
      unsigned short hv = f2bf(emb[(size_t)tok * ND + tid]);
      #pragma unroll
      for (int r = 0; r < 8; ++r) comb[r * COMB_STR + 128 + tid] = hv;
    }
    if (t + 1 < NT && tid == 0)
      __builtin_prefetch(emb + (size_t)xin[b * NT + t + 1] * ND, 0, 1);
    __syncthreads();

    // ---- B: [z|r] = sigmoid([s|x] @ Wzr^T + b) ; 16 n-tiles, 2 per wave, K=256
    AccU a0, a1;
    #pragma unroll
    for (int g = 0; g < 8; ++g) { a0.f[g] = 0.f; a1.f[g] = 0.f; }
    #pragma unroll
    for (int ks = 0; ks < 8; ++ks) {
      v16bf af = load_afrag(comb, COMB_STR, ks * 32, lane);
      v16bf b0 = load_bfrag(wzr, 2 * wave,     ks, 8, lane);
      v16bf b1 = load_bfrag(wzr, 2 * wave + 1, ks, 8, lane);
      a0.v = wmma_bf16(af, b0, a0.v);
      a1.v = wmma_bf16(af, b1, a1.v);
    }
    __syncthreads();   // everyone done reading comb before r-waves overwrite cols 0-127
    if (lane < 16) {
      #pragma unroll
      for (int p = 0; p < 2; ++p) {
        int nt = 2 * wave + p;
        int ncol = nt * 16 + lane;
        #pragma unroll
        for (int g = 0; g < 8; ++g) {
          float v = sigmoidf_((p ? a1.f[g] : a0.f[g]) + bias[ncol]);
          if (nt < 8) zb[g * 128 + ncol] = v;                         // z
          else { int c = ncol - 128;                                  // r*states -> comb
                 comb[g * COMB_STR + c] = f2bf(v * st[g * 128 + c]); }
        }
      }
    }
    __syncthreads();

    // ---- C: cand = tanh([r*s|x] @ Wh^T + bh); new = (1-z)s + z*cand ; 8 tiles, K=256
    AccU ac;
    #pragma unroll
    for (int g = 0; g < 8; ++g) ac.f[g] = 0.f;
    #pragma unroll
    for (int ks = 0; ks < 8; ++ks) {
      v16bf af = load_afrag(comb, COMB_STR, ks * 32, lane);
      v16bf bf = load_bfrag(whf, wave, ks, 8, lane);
      ac.v = wmma_bf16(af, bf, ac.v);
    }
    __syncthreads();   // done reading comb before writing bf16(new) into cols 0-127
    if (lane < 16) {
      int c = wave * 16 + lane;
      #pragma unroll
      for (int g = 0; g < 8; ++g) {
        float cand = tanhf(ac.f[g] + bias[256 + c]);
        float zv = zb[g * 128 + c];
        float nv = (1.f - zv) * st[g * 128 + c] + zv * cand;
        st[g * 128 + c] = nv;
        comb[g * COMB_STR + c] = f2bf(nv);
      }
    }
    __syncthreads();

    // ---- D: q,k,v = new @ W^T + b ; A = comb[:,0:128], 3 tiles/wave, K=128 (L2 frags)
    AccU aq, ak2, av2;
    #pragma unroll
    for (int g = 0; g < 8; ++g) { aq.f[g] = 0.f; ak2.f[g] = 0.f; av2.f[g] = 0.f; }
    #pragma unroll
    for (int ks = 0; ks < 4; ++ks) {
      v16bf af  = load_afrag(comb, COMB_STR, ks * 32, lane);
      v16bf bq2 = load_bfrag(wqkvf, wave,      ks, 4, lane);
      v16bf bk3 = load_bfrag(wqkvf, wave + 8,  ks, 4, lane);
      v16bf bv3 = load_bfrag(wqkvf, wave + 16, ks, 4, lane);
      aq.v  = wmma_bf16(af, bq2, aq.v);
      ak2.v = wmma_bf16(af, bk3, ak2.v);
      av2.v = wmma_bf16(af, bv3, av2.v);
    }
    if (lane < 16) {
      int c = wave * 16 + lane;
      #pragma unroll
      for (int g = 0; g < 8; ++g) {
        qb[g * 128 + c] = aq.f[g]  + bias[384 + c];
        kb[g * 128 + c] = ak2.f[g] + bias[512 + c];
        vb[g * 128 + c] = av2.f[g] + bias[640 + c];
      }
    }
    __syncthreads();

    // ---- E: 4-head attention over 8 state tokens (tiny -> VALU)
    {
      int h = tid >> 6, qi = (tid >> 3) & 7, ki = tid & 7;
      const float* qp = qb + qi * 128 + h * 32;
      const float* kp = kb + ki * 128 + h * 32;
      float s = 0.f;
      #pragma unroll
      for (int d2 = 0; d2 < 32; ++d2) s += qp[d2] * kp[d2];
      sc[tid] = s * 0.1767766952966369f;   // 1/sqrt(32)
    }
    __syncthreads();
    if (tid < 32) {                         // softmax per (h,q)
      float* row = sc + tid * 8;
      float m = row[0];
      #pragma unroll
      for (int i2 = 1; i2 < 8; ++i2) m = fmaxf(m, row[i2]);
      float s = 0.f;
      #pragma unroll
      for (int i2 = 0; i2 < 8; ++i2) { float e = expf(row[i2] - m); row[i2] = e; s += e; }
      float inv = 1.f / s;
      #pragma unroll
      for (int i2 = 0; i2 < 8; ++i2) row[i2] *= inv;
    }
    __syncthreads();
    for (int e = tid; e < 1024; e += 256) { // head mix -> act2 (bf16, input to Wo)
      int qi = e >> 7, c = e & 127, h = c >> 5;
      const float* arow = sc + (h * 8 + qi) * 8;
      float s = 0.f;
      #pragma unroll
      for (int k2 = 0; k2 < 8; ++k2) s += arow[k2] * vb[k2 * 128 + c];
      act2[qi * ACT_STR + c] = f2bf(s);
    }
    __syncthreads();

    // ---- F: attn = omix @ Wo^T + bo ; 8 tiles, K=128 (L2 frags)
    AccU ao;
    #pragma unroll
    for (int g = 0; g < 8; ++g) ao.f[g] = 0.f;
    #pragma unroll
    for (int ks = 0; ks < 4; ++ks) {
      v16bf af = load_afrag(act2, ACT_STR, ks * 32, lane);
      v16bf bf = load_bfrag(wof, wave, ks, 4, lane);
      ao.v = wmma_bf16(af, bf, ao.v);
    }
    if (lane < 16) {
      int c = wave * 16 + lane;
      #pragma unroll
      for (int g = 0; g < 8; ++g) {
        float avv = ao.f[g] + bias[768 + c];
        at[g * 128 + c] = avv;
        comb[g * COMB_STR + 128 + c] = f2bf(avv);   // [new|attn] for phase G
      }
    }
    __syncthreads();

    // ---- G: g = sigmoid([new|attn] @ Wc^T + bc); s = g*new + (1-g)*attn ; K=256
    AccU ag;
    #pragma unroll
    for (int g = 0; g < 8; ++g) ag.f[g] = 0.f;
    #pragma unroll
    for (int ks = 0; ks < 8; ++ks) {
      v16bf af = load_afrag(comb, COMB_STR, ks * 32, lane);
      v16bf bf = load_bfrag(wcf, wave, ks, 8, lane);
      ag.v = wmma_bf16(af, bf, ag.v);
    }
    __syncthreads();   // done reading comb before writing next-step bf16(states)
    if (lane < 16) {
      int c = wave * 16 + lane;
      #pragma unroll
      for (int g = 0; g < 8; ++g) {
        float gv  = sigmoidf_(ag.f[g] + bias[896 + c]);
        float nv  = st[g * 128 + c];
        float avv = at[g * 128 + c];
        float fs  = gv * nv + (1.f - gv) * avv;
        st[g * 128 + c] = fs;
        comb[g * COMB_STR + c] = f2bf(fs);
      }
    }
    __syncthreads();
  }

  for (int i = tid; i < 1024; i += 256) st_out[b * 1024 + i] = st[i];
}

// ---------------- final projection: out[8,32000] = states[8,1024] @ Wout^T + bout ----------------
// Pure HBM-bandwidth bound (131 MB fp32 of Wout). Wave-per-output-column dot products.
__global__ __launch_bounds__(256) void out_gemm(const float* __restrict__ stg,
                                                const float* __restrict__ Wout,
                                                const float* __restrict__ bout,
                                                float* __restrict__ out) {
  __shared__ float s[8 * 1024];
  int tid = threadIdx.x;
  for (int i = tid; i < 8192; i += 256) s[i] = stg[i];
  __syncthreads();
  int lane = tid & 31, wave = tid >> 5;
  int n = blockIdx.x * 8 + wave;
  const float* wr = Wout + (size_t)n * 1024;
  float acc[8];
  #pragma unroll
  for (int b2 = 0; b2 < 8; ++b2) acc[b2] = 0.f;
  for (int kk = 0; kk < 32; ++kk) {
    int k = lane + (kk << 5);
    float w = wr[k];
    #pragma unroll
    for (int b2 = 0; b2 < 8; ++b2) acc[b2] += w * s[b2 * 1024 + k];
  }
  #pragma unroll
  for (int b2 = 0; b2 < 8; ++b2) {
    float v = acc[b2];
    #pragma unroll
    for (int off = 16; off > 0; off >>= 1) v += __shfl_down(v, off, 32);
    if (lane == 0) out[(size_t)b2 * NOUT + n] = v + bout[n];
  }
}

extern "C" void kernel_launch(void* const* d_in, const int* in_sizes, int n_in,
                              void* d_out, int out_size, void* d_ws, size_t ws_size,
                              hipStream_t stream) {
  (void)in_sizes; (void)n_in; (void)out_size; (void)ws_size;
  const int*   x    = (const int*)d_in[0];
  const float* emb  = (const float*)d_in[1];
  const float* inis = (const float*)d_in[2];
  const float* Wz   = (const float*)d_in[3];  const float* bz   = (const float*)d_in[4];
  const float* Wr   = (const float*)d_in[5];  const float* br   = (const float*)d_in[6];
  const float* Wh   = (const float*)d_in[7];  const float* bh   = (const float*)d_in[8];
  const float* Wq   = (const float*)d_in[9];  const float* bq   = (const float*)d_in[10];
  const float* Wk   = (const float*)d_in[11]; const float* bk   = (const float*)d_in[12];
  const float* Wv   = (const float*)d_in[13]; const float* bv   = (const float*)d_in[14];
  const float* Wo   = (const float*)d_in[15]; const float* bo   = (const float*)d_in[16];
  const float* Wc   = (const float*)d_in[17]; const float* bc   = (const float*)d_in[18];
  const float* Wout = (const float*)d_in[19]; const float* bout = (const float*)d_in[20];
  float* out = (float*)d_out;

  char* ws = (char*)d_ws;
  unsigned short* wzrf  = (unsigned short*)(ws + 0);        // 131072 B
  unsigned short* whf   = (unsigned short*)(ws + 131072);   //  65536 B
  unsigned short* wcf   = (unsigned short*)(ws + 196608);   //  65536 B
  unsigned short* wqkvf = (unsigned short*)(ws + 262144);   //  98304 B
  unsigned short* wof   = (unsigned short*)(ws + 360448);   //  32768 B
  float*          stg   = (float*)(ws + 393216);            //  32768 B

  // 1) pack weights into bf16 WMMA B-frag order
  pack_frags<<<256, 256, 0, stream>>>(Wz, Wr, nullptr, 128, 256, 65536, wzrf);
  pack_frags<<<128, 256, 0, stream>>>(Wh, nullptr, nullptr, 128, 256, 32768, whf);
  pack_frags<<<128, 256, 0, stream>>>(Wc, nullptr, nullptr, 128, 256, 32768, wcf);
  pack_frags<<<192, 256, 0, stream>>>(Wq, Wk, Wv, 128, 128, 49152, wqkvf);
  pack_frags<<< 64, 256, 0, stream>>>(Wo, nullptr, nullptr, 128, 128, 16384, wof);

  // 2) persistent per-batch recurrent scan (one workgroup per WGP, big dynamic LDS)
  hipFuncSetAttribute((const void*)nsm_scan,
                      hipFuncAttributeMaxDynamicSharedMemorySize, (int)LDS_TOTAL);
  nsm_scan<<<NB, 256, LDS_TOTAL, stream>>>(x, emb, inis,
                                           (const uint4*)wzrf, wqkvf, wof,
                                           bz, br, bh, bq, bk, bv, bo, bc, stg);

  // 3) bandwidth-bound output projection
  out_gemm<<<NOUT / 8, 256, 0, stream>>>(stg, Wout, bout, out);
}
